// GraphTransformerNet_79903571575360
// MI455X (gfx1250) — compile-verified, matrix operand order checked by reading, not run on hardware
//
#include <hip/hip_runtime.h>

#define NN   20000
#define EE   320000
#define IND  256
#define HIDD 128
#define POSD 8
#define CLUS 10
#define FINL 64

typedef __attribute__((ext_vector_type(16))) __bf16 v16bf;
typedef __attribute__((ext_vector_type(8)))  __bf16 v8bf;
typedef __attribute__((ext_vector_type(8)))  float  v8f;

// ---------- helpers ----------

static __device__ __forceinline__ __bf16 f2bf(float x) {
  return (__bf16)x;
}

static __device__ __forceinline__ void atomicMaxF32(float* addr, float val) {
  if (val >= 0.0f) {
    atomicMax((int*)addr, __float_as_int(val));
  } else {
    atomicMin((unsigned int*)addr, __float_as_uint(val));
  }
}

static inline float host_bits_f32(unsigned bits) {
  union { unsigned u; float f; } c; c.u = bits; return c.f;
}

// ---------- fast WMMA GEMM (K % 128 == 0, Nrows % 16 == 0) ----------
// Y = act( X[N,K] @ W[K,M] + bias + Cin ).  Each wave computes a 16 x (CT*16)
// output strip: one A fragment (4x global_load_b128 + 8x v_cvt_pk_bf16_f32)
// feeds CT back-to-back v_wmma_f32_16x16x32_bf16, B fragments streamed from
// LDS (2x ds_load_b128 each).  W is staged per block into LDS (bf16,
// transposed to WT[col][k]) in K-chunks of 128 (<=32 KB of the 320 KB WGP LDS).
// act: 0 = none, 1 = ELU.

#define KC 128   // K-chunk staged in LDS per pass

extern __shared__ __bf16 smem_wt[];   // [strip*16][KC]

template <int CT>
__global__ __launch_bounds__(256)
void gemm_bf16_wmma_fast(const float* __restrict__ X,
                         const float* __restrict__ W,
                         const float* __restrict__ bias,
                         const float* __restrict__ Cin,
                         float* __restrict__ Y,
                         int Nrows, int K, int M, int act)
{
  const int lane = threadIdx.x & 31;
  const int wave = threadIdx.x >> 5;
  const int ntn  = (M + 15) >> 4;
  const int ntm  = Nrows >> 4;

  const int strip  = (ntn >= 8) ? 8 : ntn;      // 16-col tiles staged per block
  const int Mstage = strip * 16;                // <= 128 columns
  const int col0   = blockIdx.y * Mstage;

  const int wpr  = strip / CT;                  // waves per row-group (1 or 2)
  const int wrow = wave / wpr;
  const int wcol = wave - wrow * wpr;
  const int tmg  = blockIdx.x * (8 / wpr) + wrow;
  const int valid = (tmg < ntm);

  const int rl      = lane & 15;
  const int row     = tmg * 16 + rl;            // < Nrows when valid (N%16==0)
  const int clbase  = wcol * CT * 16 + rl;      // local column of tile 0
  const int khalf   = (lane >> 4) << 3;         // 0 / 8

  v8f acc[CT];
#pragma unroll
  for (int c = 0; c < CT; ++c) acc[c] = (v8f){0.f,0.f,0.f,0.f,0.f,0.f,0.f,0.f};

  for (int kc = 0; kc < K; kc += KC) {
    // ---- stage W[kc:kc+KC, col0:col0+Mstage] into LDS as bf16, transposed
    for (int idx = threadIdx.x; idx < Mstage * KC; idx += 256) {
      const int ml = idx / KC;
      const int kl = idx - ml * KC;
      const int k  = kc + kl;
      const int m  = col0 + ml;
      float v = 0.f;
      if (m < M) v = W[(size_t)k * M + m];      // k < K since K % KC == 0
      smem_wt[idx] = f2bf(v);
    }
    __syncthreads();

    if (valid) {
      const float* ap = X + (size_t)row * K + kc + khalf;
#pragma unroll
      for (int k0 = 0; k0 < KC; k0 += 32) {
        // A fragment: two contiguous 8-float groups -> 4x b128 loads, 8 cvts
        const float4 f0 = *(const float4*)(ap + k0);
        const float4 f1 = *(const float4*)(ap + k0 + 4);
        const float4 f2 = *(const float4*)(ap + k0 + 16);
        const float4 f3 = *(const float4*)(ap + k0 + 20);
        v16bf a;
        a[0]  = f2bf(f0.x); a[1]  = f2bf(f0.y); a[2]  = f2bf(f0.z); a[3]  = f2bf(f0.w);
        a[4]  = f2bf(f1.x); a[5]  = f2bf(f1.y); a[6]  = f2bf(f1.z); a[7]  = f2bf(f1.w);
        a[8]  = f2bf(f2.x); a[9]  = f2bf(f2.y); a[10] = f2bf(f2.z); a[11] = f2bf(f2.w);
        a[12] = f2bf(f3.x); a[13] = f2bf(f3.y); a[14] = f2bf(f3.z); a[15] = f2bf(f3.w);
        // CT column tiles reuse the A fragment from registers
#pragma unroll
        for (int c = 0; c < CT; ++c) {
          const __bf16* bp = &smem_wt[(clbase + c * 16) * KC + khalf + k0];
          const v8bf b0 = *(const v8bf*)(bp);
          const v8bf b1 = *(const v8bf*)(bp + 16);
          v16bf b;
#pragma unroll
          for (int i = 0; i < 8; ++i) { b[i] = b0[i]; b[i + 8] = b1[i]; }
          acc[c] = __builtin_amdgcn_wmma_f32_16x16x32_bf16(
              false, a, false, b, (short)0, acc[c], false, false);
        }
      }
    }
    __syncthreads();
  }

  if (!valid) return;
#pragma unroll
  for (int c = 0; c < CT; ++c) {
    const int col = col0 + wcol * CT * 16 + c * 16 + rl;
    if (col < M) {
#pragma unroll
      for (int j = 0; j < 8; ++j) {
        const int m = (lane < 16) ? j : (j + 8);
        const int r = tmg * 16 + m;
        float y = acc[c][j];
        if (bias) y += bias[col];
        if (Cin)  y += Cin[(size_t)r * M + col];
        if (act == 1) y = (y > 0.f) ? y : (__expf(y) - 1.f);
        Y[(size_t)r * M + col] = y;
      }
    }
  }
}

// ---------- fused lap projection: Y = lap[N,8] @ W[8,128] + b + Cin ----------
// K=8 is a poor fit for the K=32 WMMA (75% zero MACs); 8 VALU FMAs per output
// element with W staged in 4 KB of LDS is the right tool.  2 rows per block.

__global__ __launch_bounds__(256)
void lap_proj_k(const float* __restrict__ lapx,
                const float* __restrict__ W,
                const float* __restrict__ bias,
                const float* __restrict__ Cin,
                float* __restrict__ Y, int Nrows)
{
  __shared__ float Wl[POSD * HIDD];
  for (int i = threadIdx.x; i < POSD * HIDD; i += 256) Wl[i] = W[i];
  __syncthreads();
  const int t   = blockIdx.x * 256 + threadIdx.x;
  const int row = t >> 7;
  const int col = t & 127;
  if (row >= Nrows) return;
  const float4* lp = (const float4*)(lapx + (size_t)row * POSD);
  const float4 l0 = lp[0];
  const float4 l1 = lp[1];
  float y = bias[col] + Cin[(size_t)row * HIDD + col];
  y += l0.x * Wl[0 * HIDD + col] + l0.y * Wl[1 * HIDD + col]
     + l0.z * Wl[2 * HIDD + col] + l0.w * Wl[3 * HIDD + col];
  y += l1.x * Wl[4 * HIDD + col] + l1.y * Wl[5 * HIDD + col]
     + l1.z * Wl[6 * HIDD + col] + l1.w * Wl[7 * HIDD + col];
  Y[(size_t)row * HIDD + col] = y;
}

// ---------- elementwise fill ----------

__global__ void fill_k(float* __restrict__ p, float v, int n) {
  int i = blockIdx.x * blockDim.x + threadIdx.x;
  if (i < n) p[i] = v;
}

// ---------- edge attention ----------

__global__ void edge_logits_k(const float* __restrict__ q,
                              const float* __restrict__ kk,
                              const int* __restrict__ src,
                              const int* __restrict__ dst,
                              float* __restrict__ logit,
                              float* __restrict__ m)
{
  int t = blockIdx.x * blockDim.x + threadIdx.x;
  if (t >= EE * 8) return;
  const int e = t >> 3, h = t & 7;
  const int s = src[e], d = dst[e];
  const float4* qp = (const float4*)(q  + (size_t)d * HIDD + h * 16);
  const float4* kp = (const float4*)(kk + (size_t)s * HIDD + h * 16);
  float acc = 0.f;
#pragma unroll
  for (int i = 0; i < 4; ++i) {
    const float4 a = qp[i], b = kp[i];
    acc += a.x * b.x + a.y * b.y + a.z * b.z + a.w * b.w;
  }
  acc *= 0.25f;                                // 1/sqrt(16)
  logit[t] = acc;
  atomicMaxF32(&m[(size_t)d * 8 + h], acc);
}

__global__ void edge_exp_k(float* __restrict__ logit,
                           const float* __restrict__ m,
                           const int* __restrict__ dst,
                           float* __restrict__ den)
{
  int t = blockIdx.x * blockDim.x + threadIdx.x;
  if (t >= EE * 8) return;
  const int e = t >> 3, h = t & 7;
  const int d = dst[e];
  float mm = m[(size_t)d * 8 + h];
  mm = (mm < -1e30f) ? 0.f : mm;
  const float ex = __expf(logit[t] - mm);
  logit[t] = ex;
  atomicAdd(&den[(size_t)d * 8 + h], ex);
}

__global__ void edge_scatter_k(const float* __restrict__ ex,
                               const float* __restrict__ den,
                               const float* __restrict__ v,
                               const int* __restrict__ src,
                               const int* __restrict__ dst,
                               float* __restrict__ out)
{
  int t = blockIdx.x * blockDim.x + threadIdx.x;
  if (t >= EE * 8) return;
  const int e = t >> 3, h = t & 7;
  const int s = src[e], d = dst[e];
  const float alpha = ex[t] / (den[(size_t)d * 8 + h] + 1e-16f);
  const float4* vp = (const float4*)(v + (size_t)s * HIDD + h * 16);
  float*        op = out + (size_t)d * HIDD + h * 16;
#pragma unroll
  for (int i = 0; i < 4; ++i) {
    const float4 a = vp[i];
    atomicAdd(&op[4 * i + 0], alpha * a.x);
    atomicAdd(&op[4 * i + 1], alpha * a.y);
    atomicAdd(&op[4 * i + 2], alpha * a.z);
    atomicAdd(&op[4 * i + 3], alpha * a.w);
  }
}

// ---------- layernorm: one wave32 per row of 128 ----------

__global__ void layernorm_k(const float* __restrict__ in,
                            const float* __restrict__ g,
                            const float* __restrict__ b,
                            float* __restrict__ out, int Nrows)
{
  const int lane = threadIdx.x & 31;
  const int row  = blockIdx.x * (blockDim.x >> 5) + (threadIdx.x >> 5);
  if (row >= Nrows) return;
  const float* r = in + (size_t)row * HIDD;
  float x[4];
#pragma unroll
  for (int j = 0; j < 4; ++j) x[j] = r[lane + 32 * j];
  float s = x[0] + x[1] + x[2] + x[3];
#pragma unroll
  for (int off = 16; off > 0; off >>= 1) s += __shfl_xor(s, off, 32);
  const float mean = s * (1.f / HIDD);
  float vs = 0.f;
#pragma unroll
  for (int j = 0; j < 4; ++j) { float d = x[j] - mean; vs += d * d; }
#pragma unroll
  for (int off = 16; off > 0; off >>= 1) vs += __shfl_xor(vs, off, 32);
  const float inv = rsqrtf(vs * (1.f / HIDD) + 1e-5f);
  float* o = out + (size_t)row * HIDD;
#pragma unroll
  for (int j = 0; j < 4; ++j) {
    const int c = lane + 32 * j;
    o[c] = g[c] * ((x[j] - mean) * inv) + b[c];
  }
}

// ---------- host orchestration ----------

static void gemm(hipStream_t s, const float* X, const float* W, const float* bias,
                 const float* Cin, float* Y, int n, int k, int m, int act) {
  // all callers have k % 32 == 0 and n % 16 == 0
  const int ntn = (m + 15) / 16;
  const int ntm = (n + 15) / 16;
  const int strip = (ntn >= 8) ? 8 : ntn;
  const int CT    = (ntn >= 4) ? 4 : 1;
  const int wpr   = strip / CT;                     // 1 or 2
  const int rowsPerBlock = 8 / wpr;                 // row tiles per block
  const int gx = (ntm + rowsPerBlock - 1) / rowsPerBlock;
  const int gy = (ntn + strip - 1) / strip;
  const size_t shbytes = (size_t)strip * 16 * KC * sizeof(__bf16);  // <= 32 KB
  if (CT == 4)
    gemm_bf16_wmma_fast<4><<<dim3(gx, gy), 256, shbytes, s>>>(X, W, bias, Cin, Y, n, k, m, act);
  else
    gemm_bf16_wmma_fast<1><<<dim3(gx, gy), 256, shbytes, s>>>(X, W, bias, Cin, Y, n, k, m, act);
}

static void fill(hipStream_t s, float* p, float v, int n) {
  fill_k<<<(n + 255) / 256, 256, 0, s>>>(p, v, n);
}

struct LayerP {
  const float *Wq,*bq,*Wk,*bk,*Wv,*bv,*Wskip,*bskip,*OW,*Ob;
  const float *ln1g,*ln1b,*ln2g,*ln2b,*f1W,*f1b,*f2W,*f2b;
};

static LayerP layer_at(void* const* d_in, int base) {
  LayerP L;
  L.Wq    = (const float*)d_in[base + 0];  L.bq    = (const float*)d_in[base + 1];
  L.Wk    = (const float*)d_in[base + 2];  L.bk    = (const float*)d_in[base + 3];
  L.Wv    = (const float*)d_in[base + 4];  L.bv    = (const float*)d_in[base + 5];
  L.Wskip = (const float*)d_in[base + 6];  L.bskip = (const float*)d_in[base + 7];
  L.OW    = (const float*)d_in[base + 8];  L.Ob    = (const float*)d_in[base + 9];
  L.ln1g  = (const float*)d_in[base + 10]; L.ln1b  = (const float*)d_in[base + 11];
  L.ln2g  = (const float*)d_in[base + 12]; L.ln2b  = (const float*)d_in[base + 13];
  L.f1W   = (const float*)d_in[base + 14]; L.f1b   = (const float*)d_in[base + 15];
  L.f2W   = (const float*)d_in[base + 16]; L.f2b   = (const float*)d_in[base + 17];
  return L;
}

struct Ws {
  float *q, *k, *v, *attn, *t1, *hbig, *logit, *m, *den;
};

static void run_layer(hipStream_t s, float* x, const int* src, const int* dst,
                      const LayerP& L, const Ws& w)
{
  const int EH = EE * 8;
  gemm(s, x, L.Wq, L.bq, nullptr, w.q, NN, HIDD, HIDD, 0);
  gemm(s, x, L.Wk, L.bk, nullptr, w.k, NN, HIDD, HIDD, 0);
  gemm(s, x, L.Wv, L.bv, nullptr, w.v, NN, HIDD, HIDD, 0);
  fill(s, w.m,   host_bits_f32(0xFF800000u), NN * 8);  // -inf
  fill(s, w.den, 0.f, NN * 8);
  fill(s, w.attn, 0.f, NN * HIDD);
  edge_logits_k<<<(EH + 255) / 256, 256, 0, s>>>(w.q, w.k, src, dst, w.logit, w.m);
  edge_exp_k   <<<(EH + 255) / 256, 256, 0, s>>>(w.logit, w.m, dst, w.den);
  edge_scatter_k<<<(EH + 255) / 256, 256, 0, s>>>(w.logit, w.den, w.v, src, dst, w.attn);
  gemm(s, x, L.Wskip, L.bskip, w.attn, w.t1, NN, HIDD, HIDD, 1);
  gemm(s, w.t1, L.OW, L.Ob, x, w.attn, NN, HIDD, HIDD, 0);
  layernorm_k<<<(NN + 7) / 8, 256, 0, s>>>(w.attn, L.ln1g, L.ln1b, x, NN);
  gemm(s, x, L.f1W, L.f1b, nullptr, w.hbig, NN, HIDD, 2 * HIDD, 1);
  gemm(s, w.hbig, L.f2W, L.f2b, x, w.attn, NN, 2 * HIDD, HIDD, 0);
  layernorm_k<<<(NN + 7) / 8, 256, 0, s>>>(w.attn, L.ln2g, L.ln2b, x, NN);
}

extern "C" void kernel_launch(void* const* d_in, const int* in_sizes, int n_in,
                              void* d_out, int out_size, void* d_ws, size_t ws_size,
                              hipStream_t stream)
{
  (void)in_sizes; (void)n_in; (void)out_size; (void)ws_size;

  const float* h1  = (const float*)d_in[0];
  const float* h2  = (const float*)d_in[1];
  const float* lap = (const float*)d_in[2];
  const int*   ei  = (const int*)d_in[3];
  const int* src = ei;
  const int* dst = ei + EE;

  const float* emb1_W = (const float*)d_in[4];
  const float* emb1_b = (const float*)d_in[5];
  const float* emb2_W = (const float*)d_in[6];
  const float* emb2_b = (const float*)d_in[7];
  const float* lap1_W = (const float*)d_in[8];
  const float* lap1_b = (const float*)d_in[9];
  const float* lap2_W = (const float*)d_in[10];
  const float* lap2_b = (const float*)d_in[11];
  // layers_q: bases 12, 30 ; layers_k: bases 48, 66
  const float* cq_W = (const float*)d_in[84]; const float* cq_b = (const float*)d_in[85];
  const float* ck_W = (const float*)d_in[86]; const float* ck_b = (const float*)d_in[87];
  const float* pq_W = (const float*)d_in[88]; const float* pq_b = (const float*)d_in[89];
  const float* pk_W = (const float*)d_in[90]; const float* pk_b = (const float*)d_in[91];

  float* ws = (float*)d_ws;
  const size_t U = (size_t)NN * HIDD;          // 2,560,000 floats
  Ws w;
  w.q     = ws + 0 * U;
  w.k     = ws + 1 * U;
  w.v     = ws + 2 * U;
  w.attn  = ws + 3 * U;
  w.t1    = ws + 4 * U;
  w.hbig  = ws + 5 * U;                        // N x 256 -> 2 units
  w.logit = ws + 7 * U;                        // E*8 == 2,560,000 floats
  w.m     = ws + 8 * U;
  w.den   = ws + 8 * U + (size_t)NN * 8;

  float* out = (float*)d_out;
  float* x1  = out;                            // N x 128
  float* x2  = out + U;                        // N x 128
  float* c1  = out + 2 * U;                    // N x 64
  float* c2  = c1 + (size_t)NN * FINL;         // N x 64
  float* k1  = c2 + (size_t)NN * FINL;         // N x 10
  float* k2  = k1 + (size_t)NN * CLUS;         // N x 10

  const int lapBlocks = (NN * HIDD + 255) / 256;

  // ---- tower 1 ----
  gemm(stream, h1, emb1_W, emb1_b, nullptr, w.t1, NN, IND, HIDD, 0);
  lap_proj_k<<<lapBlocks, 256, 0, stream>>>(lap, lap1_W, lap1_b, w.t1, x1, NN);
  run_layer(stream, x1, src, dst, layer_at(d_in, 12), w);
  run_layer(stream, x1, src, dst, layer_at(d_in, 30), w);

  // ---- tower 2 ----
  gemm(stream, h2, emb2_W, emb2_b, nullptr, w.t1, NN, IND, HIDD, 0);
  lap_proj_k<<<lapBlocks, 256, 0, stream>>>(lap, lap2_W, lap2_b, w.t1, x2, NN);
  run_layer(stream, x2, src, dst, layer_at(d_in, 48), w);
  run_layer(stream, x2, src, dst, layer_at(d_in, 66), w);

  // ---- heads ----
  gemm(stream, x1, pq_W, pq_b, nullptr, c1, NN, HIDD, FINL, 0);
  gemm(stream, x2, pk_W, pk_b, nullptr, c2, NN, HIDD, FINL, 0);
  gemm(stream, x1, cq_W, cq_b, nullptr, k1, NN, HIDD, CLUS, 0);
  gemm(stream, x2, ck_W, ck_b, nullptr, k2, NN, HIDD, CLUS, 0);
}
